// AttentiveReadout_27049704030899
// MI455X (gfx1250) — matrix operand order repaired and checked by reading, not compiled, not run on hardware
//
#include <hip/hip_runtime.h>
#include <hip/hip_bf16.h>
#include <math.h>

typedef __attribute__((ext_vector_type(16))) _Float16 v16h;
typedef __attribute__((ext_vector_type(8)))  float    v8f;

#define IN_DIM  256
#define OUT_DIM 256
#define NG      1024   // num_graphs (reference constant)

// ---------------------------------------------------------------------------
// float atomic-max via monotonic integer-bit ordering
// ---------------------------------------------------------------------------
__device__ __forceinline__ void atomicMaxF(float* addr, float val) {
  if (__float_as_uint(val) >> 31) {
    atomicMin((unsigned int*)addr, __float_as_uint(val));
  } else {
    atomicMax((int*)addr, __float_as_int(val));
  }
}

// ---------------------------------------------------------------------------
// Kernel 0: init segment scratch + zero graph_emb output region
// ---------------------------------------------------------------------------
__global__ void k_init(float* __restrict__ emb, float* __restrict__ segmax,
                       float* __restrict__ denom) {
  int i = blockIdx.x * blockDim.x + threadIdx.x;
  if (i < NG * OUT_DIM) emb[i] = 0.0f;
  if (i < NG) { segmax[i] = -__builtin_huge_valf(); denom[i] = 0.0f; }
}

// ---------------------------------------------------------------------------
// Kernel 1 (fast path): single fp32 pass over h.
//  - converts h into f16 WMMA A-fragment layout (ISA 7.12.2)
//  - computes logits = h·W_attn + b_attn and per-graph max
// Block = 8 waves; wave id == K-step s; one 16-row tile per block.
// ---------------------------------------------------------------------------
__global__ void __launch_bounds__(256)
k_packA(const float* __restrict__ h,
        const long long* __restrict__ batch,
        const float* __restrict__ Wa, const float* __restrict__ ba,
        v16h* __restrict__ aPk,
        float* __restrict__ logits, float* __restrict__ segmax, int N) {
  __shared__ float red[256];
  int tid  = threadIdx.x;
  int lane = tid & 31;
  int s    = tid >> 5;          // wave == K-step
  int li   = lane & 15;
  int hi   = lane >> 4;
  int rowbase = blockIdx.x * 16;

  int m = rowbase + li;
  bool valid = (m < N);
  int mc = valid ? m : (N - 1);
  const float* hrow = h + (size_t)mc * IN_DIM;

  int r0 = s * 32 + hi * 8;
  int r1 = r0 + 16;
  float4 x0 = *(const float4*)(hrow + r0);
  float4 x1 = *(const float4*)(hrow + r0 + 4);
  float4 y0 = *(const float4*)(hrow + r1);
  float4 y1 = *(const float4*)(hrow + r1 + 4);

  float4 w0 = *(const float4*)(Wa + r0);
  float4 w1 = *(const float4*)(Wa + r0 + 4);
  float4 w2 = *(const float4*)(Wa + r1);
  float4 w3 = *(const float4*)(Wa + r1 + 4);
  float p = x0.x * w0.x + x0.y * w0.y + x0.z * w0.z + x0.w * w0.w
          + x1.x * w1.x + x1.y * w1.y + x1.z * w1.z + x1.w * w1.w
          + y0.x * w2.x + y0.y * w2.y + y0.z * w2.z + y0.w * w2.w
          + y1.x * w3.x + y1.y * w3.y + y1.z * w3.z + y1.w * w3.w;
  red[tid] = valid ? p : 0.0f;

  v16h a;
  a[0]  = (_Float16)x0.x; a[1]  = (_Float16)x0.y;
  a[2]  = (_Float16)x0.z; a[3]  = (_Float16)x0.w;
  a[4]  = (_Float16)x1.x; a[5]  = (_Float16)x1.y;
  a[6]  = (_Float16)x1.z; a[7]  = (_Float16)x1.w;
  a[8]  = (_Float16)y0.x; a[9]  = (_Float16)y0.y;
  a[10] = (_Float16)y0.z; a[11] = (_Float16)y0.w;
  a[12] = (_Float16)y1.x; a[13] = (_Float16)y1.y;
  a[14] = (_Float16)y1.z; a[15] = (_Float16)y1.w;
  aPk[((size_t)blockIdx.x * 8 + s) * 32 + lane] = a;

  __syncthreads();
  if (tid < 16) {
    int mg = rowbase + tid;
    if (mg < N) {
      float sum = 0.0f;
      #pragma unroll
      for (int w = 0; w < 8; ++w)
        sum += red[w * 32 + tid] + red[w * 32 + 16 + tid];
      sum += ba[0];
      logits[mg] = sum;
      atomicMaxF(&segmax[(int)batch[mg]], sum);
    }
  }
}

// ---------------------------------------------------------------------------
// Kernel 1 (fallback path, small ws): wave-per-node logits only
// ---------------------------------------------------------------------------
__global__ void k_logits(const float* __restrict__ h,
                         const long long* __restrict__ batch,
                         const float* __restrict__ Wa,
                         const float* __restrict__ ba,
                         float* __restrict__ logits,
                         float* __restrict__ segmax, int N) {
  int lane = threadIdx.x & 31;
  int node = blockIdx.x * (blockDim.x >> 5) + (threadIdx.x >> 5);
  if (node >= N) return;
  const float4* hp = (const float4*)(h + (size_t)node * IN_DIM + lane * 8);
  const float4* wp = (const float4*)(Wa + lane * 8);
  float4 h0 = hp[0], h1 = hp[1], w0 = wp[0], w1 = wp[1];
  float p = h0.x * w0.x + h0.y * w0.y + h0.z * w0.z + h0.w * w0.w
          + h1.x * w1.x + h1.y * w1.y + h1.z * w1.z + h1.w * w1.w;
  #pragma unroll
  for (int m = 16; m; m >>= 1) p += __shfl_xor(p, m, 32);
  if (lane == 0) {
    float lg = p + ba[0];
    logits[node] = lg;
    atomicMaxF(&segmax[(int)batch[node]], lg);
  }
}

// ---------------------------------------------------------------------------
// Kernel 2: e = exp(logit - segmax[b]) (in place) ; denom[b] += e
// ---------------------------------------------------------------------------
__global__ void k_expsum(const long long* __restrict__ batch,
                         const float* __restrict__ segmax,
                         float* __restrict__ logits,
                         float* __restrict__ denom, int N) {
  int n = blockIdx.x * blockDim.x + threadIdx.x;
  if (n >= N) return;
  int b = (int)batch[n];
  float e = __expf(logits[n] - segmax[b]);
  logits[n] = e;
  atomicAdd(&denom[b], e);
}

// ---------------------------------------------------------------------------
// Kernel 3: weights[n] = e / denom[b]  (written straight to d_out tail)
// ---------------------------------------------------------------------------
__global__ void k_weights(const long long* __restrict__ batch,
                          const float* __restrict__ logits,
                          const float* __restrict__ denom,
                          float* __restrict__ wout, int N) {
  int n = blockIdx.x * blockDim.x + threadIdx.x;
  if (n >= N) return;
  wout[n] = logits[n] / denom[(int)batch[n]];
}

// ---------------------------------------------------------------------------
// Kernel 4: prepack W_gate/W_value into f16 WMMA B-fragments
// ---------------------------------------------------------------------------
__global__ void k_pack(const float* __restrict__ Wg,
                       const float* __restrict__ Wv,
                       v16h* __restrict__ pk) {
  int t = blockIdx.x * blockDim.x + threadIdx.x;   // 256 frags * 32 lanes
  int f = t >> 5, l = t & 31;
  int mat = f >> 7;
  int ct  = (f >> 3) & 15;
  int s   = f & 7;
  const float* W = mat ? Wv : Wg;
  int n  = ct * 16 + (l & 15);
  int kb = s * 32 + ((l >> 4) << 3);
  v16h o;
  #pragma unroll
  for (int j = 0; j < 16; ++j)
    o[j] = (_Float16)W[(size_t)(kb + j + ((j >> 3) << 3)) * OUT_DIM + n];
  pk[f * 32 + l] = o;
}

// ---------------------------------------------------------------------------
// Shared epilogue: bias + sigmoid gate + attention weight + segment sum.
// ---------------------------------------------------------------------------
__device__ __forceinline__ void
epilogue_tile(const long long* __restrict__ batch,
              const float* __restrict__ wgt, float* __restrict__ emb,
              int N, int rowbase, int li, int hi,
              int ct0, int ct1,
              float bg0c, float bv0c, float bg1c, float bv1c,
              const v8f& ag0, const v8f& av0,
              const v8f& ag1, const v8f& av1) {
  float w8[8]; int g8[8];
  #pragma unroll
  for (int r = 0; r < 8; ++r) {
    int m = rowbase + hi * 8 + r;
    if (m < N) { w8[r] = wgt[m]; g8[r] = (int)batch[m]; }
    else       { w8[r] = 0.0f;   g8[r] = 0; }
  }
  int  gfirst  = (int)batch[rowbase];
  int  lastm   = rowbase + 15;
  bool full    = lastm < N;
  int  glast   = (int)batch[full ? lastm : (N - 1)];
  bool uniform = full && (gfirst == glast);

  auto one = [&](int ct, float bgc, float bvc, const v8f& ag, const v8f& av) {
    int n = ct * 16 + li;
    float partial = 0.0f;
    #pragma unroll
    for (int r = 0; r < 8; ++r) {
      int m = rowbase + hi * 8 + r;
      if (m >= N) continue;
      float g = ag[r] + bgc;
      float v = av[r] + bvc;
      float o = v * (1.0f / (1.0f + __expf(-g))) * w8[r];
      if (uniform) partial += o;
      else atomicAdd(&emb[(size_t)g8[r] * OUT_DIM + n], o);
    }
    if (uniform) {
      partial += __shfl_xor(partial, 16, 32);
      if (hi == 0) atomicAdd(&emb[(size_t)gfirst * OUT_DIM + n], partial);
    }
  };
  one(ct0, bg0c, bv0c, ag0, av0);
  one(ct1, bg1c, bv1c, ag1, av1);
}

// ---------------------------------------------------------------------------
// Kernel 5a (fast path): persistent GEMM, B-fragments register-resident.
// Each wave owns col-tiles {w, w+8} for gate+value: 32 v16h of B loaded ONCE,
// then a grid-stride loop over 16-row tiles: 8 A loads + 32 WMMAs + epilogue.
// ---------------------------------------------------------------------------
__global__ void __launch_bounds__(256)
k_gemm_pk(const long long* __restrict__ batch,
          const v16h* __restrict__ aPk,
          const v16h* __restrict__ pk,
          const float* __restrict__ bgate,
          const float* __restrict__ bvalue,
          const float* __restrict__ wgt,
          float* __restrict__ emb, int N, int numTiles) {
  int tid  = threadIdx.x;
  int lane = tid & 31;
  int wv   = tid >> 5;
  int li   = lane & 15;
  int hi   = lane >> 4;
  int ct0 = wv, ct1 = wv + 8;

  // B fragments: resident for the whole kernel (32 x v16h = 256 VGPRs)
  v16h Bg0[8], Bg1[8], Bv0[8], Bv1[8];
  #pragma unroll
  for (int s = 0; s < 8; ++s) {
    Bg0[s] = pk[(size_t)((( 0 + ct0) * 8 + s)) * 32 + lane];
    Bg1[s] = pk[(size_t)((( 0 + ct1) * 8 + s)) * 32 + lane];
    Bv0[s] = pk[(size_t)(((16 + ct0) * 8 + s)) * 32 + lane];
    Bv1[s] = pk[(size_t)(((16 + ct1) * 8 + s)) * 32 + lane];
  }
  // per-wave bias scalars (uniform over tiles)
  float bg0c = bgate [ct0 * 16 + li], bv0c = bvalue[ct0 * 16 + li];
  float bg1c = bgate [ct1 * 16 + li], bv1c = bvalue[ct1 * 16 + li];

  for (int t = blockIdx.x; t < numTiles; t += gridDim.x) {
    const v16h* aBase = aPk + (size_t)t * 8 * 32 + lane;
    v8f ag0 = {}, ag1 = {}, av0 = {}, av1 = {};
    #pragma unroll
    for (int s = 0; s < 8; ++s) {
      v16h a = aBase[s * 32];
      ag0 = __builtin_amdgcn_wmma_f32_16x16x32_f16(false, a, false, Bg0[s],
                                                   (short)0, ag0, false, false);
      ag1 = __builtin_amdgcn_wmma_f32_16x16x32_f16(false, a, false, Bg1[s],
                                                   (short)0, ag1, false, false);
      av0 = __builtin_amdgcn_wmma_f32_16x16x32_f16(false, a, false, Bv0[s],
                                                   (short)0, av0, false, false);
      av1 = __builtin_amdgcn_wmma_f32_16x16x32_f16(false, a, false, Bv1[s],
                                                   (short)0, av1, false, false);
    }
    epilogue_tile(batch, wgt, emb, N, t * 16, li, hi, ct0, ct1,
                  bg0c, bv0c, bg1c, bv1c, ag0, av0, ag1, av1);
  }
}

// ---------------------------------------------------------------------------
// Kernel 5b (fallback): GEMM reading fp32 h directly with on-the-fly cvt
// ---------------------------------------------------------------------------
__global__ void __launch_bounds__(256)
k_gemm(const float* __restrict__ h,
       const long long* __restrict__ batch,
       const v16h* __restrict__ pk,
       const float* __restrict__ bgate,
       const float* __restrict__ bvalue,
       const float* __restrict__ wgt,
       float* __restrict__ emb, int N) {
  int tid  = threadIdx.x;
  int lane = tid & 31;
  int wv   = tid >> 5;
  int li   = lane & 15;
  int hi   = lane >> 4;
  int rowbase = blockIdx.x * 16;
  int ct0 = wv, ct1 = wv + 8;

  int m0 = rowbase + li;
  if (m0 >= N) m0 = N - 1;
  const float* hrow = h + (size_t)m0 * IN_DIM;

  v8f ag0 = {}, ag1 = {}, av0 = {}, av1 = {};

  #pragma unroll 2
  for (int s = 0; s < 8; ++s) {
    int r0 = s * 32 + hi * 8;
    int r1 = r0 + 16;
    float4 x0 = *(const float4*)(hrow + r0);
    float4 x1 = *(const float4*)(hrow + r0 + 4);
    float4 y0 = *(const float4*)(hrow + r1);
    float4 y1 = *(const float4*)(hrow + r1 + 4);
    v16h a;
    a[0]  = (_Float16)x0.x; a[1]  = (_Float16)x0.y;
    a[2]  = (_Float16)x0.z; a[3]  = (_Float16)x0.w;
    a[4]  = (_Float16)x1.x; a[5]  = (_Float16)x1.y;
    a[6]  = (_Float16)x1.z; a[7]  = (_Float16)x1.w;
    a[8]  = (_Float16)y0.x; a[9]  = (_Float16)y0.y;
    a[10] = (_Float16)y0.z; a[11] = (_Float16)y0.w;
    a[12] = (_Float16)y1.x; a[13] = (_Float16)y1.y;
    a[14] = (_Float16)y1.z; a[15] = (_Float16)y1.w;
    v16h bg0 = pk[(size_t)((( 0 + ct0) * 8 + s)) * 32 + lane];
    v16h bg1 = pk[(size_t)((( 0 + ct1) * 8 + s)) * 32 + lane];
    v16h bv0 = pk[(size_t)(((16 + ct0) * 8 + s)) * 32 + lane];
    v16h bv1 = pk[(size_t)(((16 + ct1) * 8 + s)) * 32 + lane];
    ag0 = __builtin_amdgcn_wmma_f32_16x16x32_f16(false, a, false, bg0,
                                                 (short)0, ag0, false, false);
    ag1 = __builtin_amdgcn_wmma_f32_16x16x32_f16(false, a, false, bg1,
                                                 (short)0, ag1, false, false);
    av0 = __builtin_amdgcn_wmma_f32_16x16x32_f16(false, a, false, bv0,
                                                 (short)0, av0, false, false);
    av1 = __builtin_amdgcn_wmma_f32_16x16x32_f16(false, a, false, bv1,
                                                 (short)0, av1, false, false);
  }
  float bg0c = bgate [ct0 * 16 + li], bv0c = bvalue[ct0 * 16 + li];
  float bg1c = bgate [ct1 * 16 + li], bv1c = bvalue[ct1 * 16 + li];
  epilogue_tile(batch, wgt, emb, N, rowbase, li, hi, ct0, ct1,
                bg0c, bv0c, bg1c, bv1c, ag0, av0, ag1, av1);
}

// ---------------------------------------------------------------------------
extern "C" void kernel_launch(void* const* d_in, const int* in_sizes, int n_in,
                              void* d_out, int out_size, void* d_ws,
                              size_t ws_size, hipStream_t stream) {
  (void)n_in; (void)out_size;
  const float*     h     = (const float*)d_in[0];
  const long long* batch = (const long long*)d_in[1];
  // d_in[2] = num_graphs (compile-time NG)
  const float* Wa = (const float*)d_in[3];
  const float* ba = (const float*)d_in[4];
  const float* Wg = (const float*)d_in[5];
  const float* bg = (const float*)d_in[6];
  const float* Wv = (const float*)d_in[7];
  const float* bv = (const float*)d_in[8];

  int N = in_sizes[0] / IN_DIM;
  int numTiles = (N + 15) / 16;

  float* emb  = (float*)d_out;                 // [NG, OUT_DIM]
  float* wout = (float*)d_out + NG * OUT_DIM;  // [N] attention weights

  // workspace: logits/e (N f32) | segmax | denom | packed B | packed A (opt)
  char*  wsb    = (char*)d_ws;
  float* logits = (float*)wsb;
  size_t off    = ((size_t)N * sizeof(float) + 255) & ~(size_t)255;
  float* segmax = (float*)(wsb + off); off += 4096;
  float* denom  = (float*)(wsb + off); off += 4096;
  v16h*  pkB    = (v16h*)(wsb + off);  off += 256 * 32 * sizeof(v16h);
  v16h*  pkA    = (v16h*)(wsb + off);
  size_t need   = off + (size_t)numTiles * 8 * 32 * sizeof(v16h);
  bool fast     = (ws_size >= need);

  k_init <<<(NG * OUT_DIM + 255) / 256, 256, 0, stream>>>(emb, segmax, denom);
  k_pack <<<(256 * 32) / 256,           256, 0, stream>>>(Wg, Wv, pkB);

  if (fast) {
    k_packA<<<numTiles, 256, 0, stream>>>(h, batch, Wa, ba, pkA,
                                          logits, segmax, N);
  } else {
    k_logits<<<(N + 7) / 8, 256, 0, stream>>>(h, batch, Wa, ba,
                                              logits, segmax, N);
  }

  k_expsum <<<(N + 255) / 256, 256, 0, stream>>>(batch, segmax, logits,
                                                 denom, N);
  k_weights<<<(N + 255) / 256, 256, 0, stream>>>(batch, logits, denom,
                                                 wout, N);

  if (fast) {
    int G = numTiles < 1280 ? numTiles : 1280;   // persistent blocks
    k_gemm_pk<<<G, 256, 0, stream>>>(batch, pkA, pkB, bg, bv,
                                     wout, emb, N, numTiles);
  } else {
    k_gemm<<<numTiles, 256, 0, stream>>>(h, batch, pkB, bg, bv,
                                         wout, emb, N);
  }
}
